// _PointnetSAModuleBase_60009283059728
// MI455X (gfx1250) — matrix operand order repaired and checked by reading, not compiled, not run on hardware
//
#include <hip/hip_runtime.h>

typedef __attribute__((ext_vector_type(16))) _Float16 v16h;
typedef __attribute__((ext_vector_type(8)))  _Float16 v8h;
typedef __attribute__((ext_vector_type(8)))  float    v8f;

#define NB   8          // batches
#define NP   8192       // points per batch
#define NC   64         // feature channels
#define NS   1024       // npoint (centroids)
#define NK   32         // nsample
#define R2   0.04f      // radius^2

// ---------------------------------------------------------------- helpers
__device__ __forceinline__ v16h frag_cat(v8h lo, v8h hi) {
  return __builtin_shufflevector(lo, hi, 0,1,2,3,4,5,6,7,8,9,10,11,12,13,14,15);
}

// A fragment (16x32 f16, MxK) from an LDS tile (row-major, strideK halves).
// lane 0-15: row=lane, K = kt*32 + {0..7, 16..23}; lane 16-31: row=lane-16, K += 8.
__device__ __forceinline__ v16h load_a_lds(const _Float16* tile, int lane, int kt, int strideK) {
  int r  = lane & 15;
  int kh = (lane >> 4) * 8;
  const _Float16* p = tile + r * strideK + kt * 32 + kh;
  v8h lo = *(const v8h*)p;          // K = kt*32 + kh + 0..7
  v8h hi = *(const v8h*)(p + 16);   // K = kt*32 + 16 + kh + 0..7
  return frag_cat(lo, hi);
}

// B fragment (32x16 f16, KxN) from global weights packed column-major
// (wp[n*Kdim + k]). lane 0-15: col=lane, K = kt*32 + {0..7, 16..23}; lane 16-31: K += 8.
__device__ __forceinline__ v16h load_b_glob(const _Float16* wp, int lane, int nt, int kt, int Kdim) {
  int col = lane & 15;
  int kh  = (lane >> 4) * 8;
  const _Float16* p = wp + (nt * 16 + col) * Kdim + kt * 32 + kh;
  v8h lo = *(const v8h*)p;
  v8h hi = *(const v8h*)(p + 16);
  return frag_cat(lo, hi);
}

__device__ __forceinline__ v8f wmma16x16x32(v16h a, v16h b, v8f c) {
  return __builtin_amdgcn_wmma_f32_16x16x32_f16(false, a, false, b, (short)0, c, false, false);
}

// first-occurrence argmax combine: prefer larger value, then smaller index
__device__ __forceinline__ void amax_combine(float& bv, int& bi, float ov, int oi) {
  if (ov > bv || (ov == bv && oi < bi)) { bv = ov; bi = oi; }
}

// ---------------------------------------------------------------- kernel 1: FPS
// One workgroup (32 waves) per batch. dist[] in LDS. Per-iteration argmax:
// wave-local shuffle reduction (no barriers) -> 32 candidates -> wave 0
// shuffle-reduces those. Only 2 __syncthreads per iteration.
__global__ __launch_bounds__(1024)
void fps_kernel(const float* __restrict__ xyz, float* __restrict__ newxyz) {
  __shared__ float dist[NP];
  __shared__ float wval[32];
  __shared__ int   widx[32];
  __shared__ int   sfar;
  const int b    = blockIdx.x;
  const int t    = threadIdx.x;
  const int lane = t & 31;
  const int wv   = t >> 5;
  const float* X = xyz + (size_t)b * NP * 3;

  for (int p = t; p < NP; p += 1024) dist[p] = 1e10f;
  __syncthreads();

  int far = 0;
  for (int it = 0; it < NS; ++it) {
    if (t == 0) {
      float* o = newxyz + ((size_t)b * NS + it) * 3;
      o[0] = X[far * 3 + 0]; o[1] = X[far * 3 + 1]; o[2] = X[far * 3 + 2];
    }
    const float cx = X[far * 3 + 0], cy = X[far * 3 + 1], cz = X[far * 3 + 2];

    // per-thread scan over 8 strided points (ascending p keeps first occurrence)
    float bv = -1.0f; int bi = 0;
    for (int p = t; p < NP; p += 1024) {
      float dx = X[p * 3 + 0] - cx, dy = X[p * 3 + 1] - cy, dz = X[p * 3 + 2] - cz;
      float d  = dx * dx + dy * dy + dz * dz;
      float dm = fminf(dist[p], d);
      dist[p] = dm;
      if (dm > bv) { bv = dm; bi = p; }
    }
    // wave-local reduction via cross-lane shuffles (barrier-free)
    #pragma unroll
    for (int off = 16; off > 0; off >>= 1) {
      float ov = __shfl_xor(bv, off, 32);
      int   oi = __shfl_xor(bi, off, 32);
      amax_combine(bv, bi, ov, oi);
    }
    if (lane == 0) { wval[wv] = bv; widx[wv] = bi; }
    __syncthreads();

    // wave 0 reduces the 32 per-wave candidates
    if (wv == 0) {
      float v = wval[lane]; int i = widx[lane];
      #pragma unroll
      for (int off = 16; off > 0; off >>= 1) {
        float ov = __shfl_xor(v, off, 32);
        int   oi = __shfl_xor(i, off, 32);
        amax_combine(v, i, ov, oi);
      }
      if (lane == 0) sfar = i;
    }
    __syncthreads();
    far = sfar;
  }
}

// ---------------------------------------------------------------- kernel 2: weight pack
// f32 -> f16, column-major (B-fragment friendly), w1 zero-padded K:67->96.
__global__ void pack_weights(const float* __restrict__ w1, const float* __restrict__ w2,
                             const float* __restrict__ w3,
                             _Float16* __restrict__ w1p, _Float16* __restrict__ w2p,
                             _Float16* __restrict__ w3p) {
  int i = blockIdx.x * blockDim.x + threadIdx.x;
  if (i < 64 * 96) {
    int n = i / 96, k = i % 96;
    w1p[i] = (k < 67) ? (_Float16)w1[k * 64 + n] : (_Float16)0.0f;
  } else if (i < 64 * 96 + 64 * 64) {
    int j = i - 64 * 96; int n = j / 64, k = j % 64;
    w2p[j] = (_Float16)w2[k * 64 + n];
  } else if (i < 64 * 96 + 64 * 64 + 128 * 64) {
    int j = i - 64 * 96 - 64 * 64; int n = j / 64, k = j % 64;
    w3p[j] = (_Float16)w3[k * 128 + n];
  }
}

// ---------------------------------------------------------------- kernel 3: ball query
// One wave32 per centroid; ascending scan == reference's sorted-index semantics.
__global__ __launch_bounds__(256)
void ballq_kernel(const float* __restrict__ xyz, const float* __restrict__ newxyz,
                  int* __restrict__ nbr) {
  const int lane = threadIdx.x & 31;
  const int w    = threadIdx.x >> 5;
  const int gid  = blockIdx.x * 8 + w;       // b*NS + s
  const int b    = gid >> 10;
  const float* X = xyz + (size_t)b * NP * 3;
  const float cx = newxyz[gid * 3 + 0], cy = newxyz[gid * 3 + 1], cz = newxyz[gid * 3 + 2];
  int* out = nbr + (size_t)gid * NK;

  int cnt = 0, first = -1;
  for (int base = 0; base < NP && cnt < NK; base += 32) {
    int p = base + lane;
    float dx = X[p * 3 + 0] - cx, dy = X[p * 3 + 1] - cy, dz = X[p * 3 + 2] - cz;
    float d  = dx * dx + dy * dy + dz * dz;
    bool hit = (d <= R2);
    unsigned int mask = (unsigned int)__ballot(hit);
    if (mask) {
      if (cnt == 0) first = base + __builtin_ctz(mask);
      if (hit) {
        int slot = cnt + __popc(mask & ((1u << lane) - 1u));
        if (slot < NK) out[slot] = p;
      }
      cnt += __popc(mask);
    }
  }
  if (first < 0) first = 0;
  if (lane >= cnt && lane < NK) out[lane] = first;   // pad with first hit
}

// ---------------------------------------------------------------- kernel 4: gather + MLP + max (WMMA)
// One wave per centroid, 4 waves/block. x tile (32x96 f16) and hidden tile
// (32x64 f16) staged in LDS; 3 GEMM layers via v_wmma_f32_16x16x32_f16.
__global__ __launch_bounds__(128)
void mlp_kernel(const float* __restrict__ xyz, const float* __restrict__ feat,
                const float* __restrict__ newxyz, const int* __restrict__ nbr,
                const _Float16* __restrict__ w1p, const float* __restrict__ b1,
                const _Float16* __restrict__ w2p, const float* __restrict__ b2,
                const _Float16* __restrict__ w3p, const float* __restrict__ b3,
                float* __restrict__ newfeat) {
  __shared__ _Float16 xbuf[4][32 * 96];
  __shared__ _Float16 hbuf[4][32 * 64];
  const int lane = threadIdx.x & 31;
  const int w    = threadIdx.x >> 5;
  const int gid  = blockIdx.x * 4 + w;   // b*NS + s
  const int b    = gid >> 10;
  const int s    = gid & 1023;

  _Float16* xt = xbuf[w];
  _Float16* ht = hbuf[w];
  const v8f vzero = {0.f, 0.f, 0.f, 0.f, 0.f, 0.f, 0.f, 0.f};

  // ---- gather: lane fills row `lane` of the 32x96 input tile
  {
    int idx = nbr[(size_t)gid * NK + lane];
    float cx = newxyz[gid * 3 + 0], cy = newxyz[gid * 3 + 1], cz = newxyz[gid * 3 + 2];
    const float* P = xyz  + ((size_t)b * NP + idx) * 3;
    const float* F = feat + ((size_t)b * NP + idx) * NC;
    _Float16* row = xt + lane * 96;
    row[0] = (_Float16)(P[0] - cx);
    row[1] = (_Float16)(P[1] - cy);
    row[2] = (_Float16)(P[2] - cz);
    #pragma unroll
    for (int c = 0; c < NC; ++c) row[3 + c] = (_Float16)F[c];
    #pragma unroll
    for (int c = 67; c < 96; ++c) row[c] = (_Float16)0.0f;
  }
  __syncthreads();

  const int colL = lane & 15;
  const int rsub = (lane >> 4) * 8;

  // ---- layer 1: [32x96] @ [96x64] -> relu -> hbuf
  {
    v8f acc[2][4];
    #pragma unroll
    for (int m = 0; m < 2; ++m)
      #pragma unroll
      for (int n = 0; n < 4; ++n) acc[m][n] = vzero;
    #pragma unroll
    for (int kt = 0; kt < 3; ++kt) {
      v16h a0 = load_a_lds(xt,           lane, kt, 96);
      v16h a1 = load_a_lds(xt + 16 * 96, lane, kt, 96);
      #pragma unroll
      for (int n = 0; n < 4; ++n) {
        v16h bf = load_b_glob(w1p, lane, n, kt, 96);
        acc[0][n] = wmma16x16x32(a0, bf, acc[0][n]);
        acc[1][n] = wmma16x16x32(a1, bf, acc[1][n]);
      }
    }
    #pragma unroll
    for (int m = 0; m < 2; ++m) {
      int rbase = m * 16 + rsub;
      #pragma unroll
      for (int n = 0; n < 4; ++n) {
        int col = n * 16 + colL;
        float bias = b1[col];
        #pragma unroll
        for (int r = 0; r < 8; ++r)
          ht[(rbase + r) * 64 + col] = (_Float16)fmaxf(acc[m][n][r] + bias, 0.0f);
      }
    }
  }

  // ---- layer 2: [32x64] @ [64x64] -> relu -> hbuf (all loads precede stores)
  {
    v8f acc[2][4];
    #pragma unroll
    for (int m = 0; m < 2; ++m)
      #pragma unroll
      for (int n = 0; n < 4; ++n) acc[m][n] = vzero;
    #pragma unroll
    for (int kt = 0; kt < 2; ++kt) {
      v16h a0 = load_a_lds(ht,           lane, kt, 64);
      v16h a1 = load_a_lds(ht + 16 * 64, lane, kt, 64);
      #pragma unroll
      for (int n = 0; n < 4; ++n) {
        v16h bf = load_b_glob(w2p, lane, n, kt, 64);
        acc[0][n] = wmma16x16x32(a0, bf, acc[0][n]);
        acc[1][n] = wmma16x16x32(a1, bf, acc[1][n]);
      }
    }
    #pragma unroll
    for (int m = 0; m < 2; ++m) {
      int rbase = m * 16 + rsub;
      #pragma unroll
      for (int n = 0; n < 4; ++n) {
        int col = n * 16 + colL;
        float bias = b2[col];
        #pragma unroll
        for (int r = 0; r < 8; ++r)
          ht[(rbase + r) * 64 + col] = (_Float16)fmaxf(acc[m][n][r] + bias, 0.0f);
      }
    }
  }

  // ---- layer 3: [32x64] @ [64x128] -> relu -> max over 32 samples -> (B,128,NS)
  {
    v16h A0[2], A1[2];
    #pragma unroll
    for (int kt = 0; kt < 2; ++kt) {
      A0[kt] = load_a_lds(ht,           lane, kt, 64);
      A1[kt] = load_a_lds(ht + 16 * 64, lane, kt, 64);
    }
    float* outcol = newfeat + (size_t)b * (128 * NS) + s;
    #pragma unroll
    for (int nt = 0; nt < 8; ++nt) {
      v8f c0 = vzero, c1 = vzero;
      #pragma unroll
      for (int kt = 0; kt < 2; ++kt) {
        v16h bf = load_b_glob(w3p, lane, nt, kt, 64);
        c0 = wmma16x16x32(A0[kt], bf, c0);
        c1 = wmma16x16x32(A1[kt], bf, c1);
      }
      int col = nt * 16 + colL;
      float mx = -1e30f;
      #pragma unroll
      for (int r = 0; r < 8; ++r) mx = fmaxf(mx, fmaxf(c0[r], c1[r]));
      mx = fmaxf(mx + b3[col], 0.0f);          // bias uniform per column: add after max OK
      mx = fmaxf(mx, __shfl_xor(mx, 16, 32));  // merge lane halves (rows split across halves)
      if (lane < 16) outcol[(size_t)col * NS] = mx;
    }
  }
}

// ---------------------------------------------------------------- launch
extern "C" void kernel_launch(void* const* d_in, const int* in_sizes, int n_in,
                              void* d_out, int out_size, void* d_ws, size_t ws_size,
                              hipStream_t stream) {
  const float* xyz  = (const float*)d_in[0];
  const float* feat = (const float*)d_in[1];
  const float* w1   = (const float*)d_in[2];
  const float* b1   = (const float*)d_in[3];
  const float* w2   = (const float*)d_in[4];
  const float* b2   = (const float*)d_in[5];
  const float* w3   = (const float*)d_in[6];
  const float* b3   = (const float*)d_in[7];

  float* out      = (float*)d_out;
  float* newxyz   = out;                       // (8,1024,3)
  float* newfeat  = out + (size_t)NB * NS * 3; // (8,128,1024)

  char* ws = (char*)d_ws;
  int*      nbr = (int*)ws;                                  // 8*1024*32 ints = 1 MB
  _Float16* w1p = (_Float16*)(ws + (size_t)NB * NS * NK * 4);
  _Float16* w2p = w1p + 64 * 96;
  _Float16* w3p = w2p + 64 * 64;

  fps_kernel<<<NB, 1024, 0, stream>>>(xyz, newxyz);
  pack_weights<<<(64 * 96 + 64 * 64 + 128 * 64 + 255) / 256, 256, 0, stream>>>(
      w1, w2, w3, w1p, w2p, w3p);
  ballq_kernel<<<(NB * NS) / 8, 256, 0, stream>>>(xyz, newxyz, nbr);
  mlp_kernel<<<(NB * NS) / 4, 128, 0, stream>>>(xyz, feat, newxyz, nbr,
                                                w1p, b1, w2p, b2, w3p, b3, newfeat);
}
